// AnchorTarget_22187801051524
// MI455X (gfx1250) — compile-verified, matrix operand order checked by reading, not run on hardware
//
#include <hip/hip_runtime.h>
#include <hip/hip_bf16.h>
#include <math.h>

#define SCORE_SZ 25
#define A_NUM 5
#define N_ANCH (SCORE_SZ * SCORE_SZ * A_NUM)   // 3125
#define G_NUM 64
#define B_NUM 128
#define THR_HIGH 0.6f
#define THR_LOW 0.3f
#define POS_CAP 16
#define TOTAL_CAP 64
#define PASSB_THREADS 256

typedef __attribute__((ext_vector_type(16))) _Float16 v16h;
typedef __attribute__((ext_vector_type(8)))  float    v8f;

__device__ __forceinline__ float iou_fn(float a0, float a1, float a2, float a3,
                                        float area_a,
                                        float g0, float g1, float g2, float g3) {
    float gw = g2 - g0 + 1.0f;
    float gh = g3 - g1 + 1.0f;
    float area_g = gw * gh;
    float ix = fminf(a2, g2) - fmaxf(a0, g0) + 1.0f;
    float iy = fminf(a3, g3) - fmaxf(a1, g1) + 1.0f;
    float inter = fmaxf(ix, 0.0f) * fmaxf(iy, 0.0f);
    return inter / (area_a + area_g - inter);
}

// ---------------------------------------------------------------------------
// Pass A: per (b,n) IoU max/argmax + per-(b,g) max. One block per batch.
// gt boxes staged into LDS with CDNA5 async-load-to-LDS (ASYNCcnt path).
// ---------------------------------------------------------------------------
__global__ void __launch_bounds__(256)
passA(const float* __restrict__ gt, const float* __restrict__ anchors,
      float* __restrict__ max_ov_out, int* __restrict__ arg_ws,
      float* __restrict__ gtmax_ws) {
    __shared__ float    s_gt[G_NUM * 4];
    __shared__ unsigned s_gtmax[G_NUM];
    const int b = blockIdx.x;
    const int tid = threadIdx.x;

    // 256 threads x 1 dword = 64 boxes * 4 floats: async global -> LDS
    {
        const float* src = gt + (size_t)b * G_NUM * 4 + tid;
        unsigned ldsoff = (unsigned)(size_t)(&s_gt[tid]);   // low 32 bits = LDS addr
        asm volatile("global_load_async_to_lds_b32 %0, %1, off"
                     :: "v"(ldsoff), "v"(src) : "memory");
        asm volatile("s_wait_asynccnt 0" ::: "memory");
    }
    if (tid < G_NUM) s_gtmax[tid] = 0u;
    __syncthreads();

    for (int n = tid; n < N_ANCH; n += blockDim.x) {
        float4 an = ((const float4*)anchors)[n];
        float aw = an.z - an.x + 1.0f;
        float ah = an.w - an.y + 1.0f;
        float area_a = aw * ah;
        float best = -1.0f;
        int bg = 0;
        for (int g = 0; g < G_NUM; ++g) {
            float v = iou_fn(an.x, an.y, an.z, an.w, area_a,
                             s_gt[g * 4 + 0], s_gt[g * 4 + 1],
                             s_gt[g * 4 + 2], s_gt[g * 4 + 3]);
            if (v > best) { best = v; bg = g; }        // first-max like argmax
            atomicMax(&s_gtmax[g], __float_as_uint(v)); // IoU >= 0 -> uint cmp ok
        }
        max_ov_out[(size_t)b * N_ANCH + n] = best;
        arg_ws[(size_t)b * N_ANCH + n] = bg;
    }
    __syncthreads();
    if (tid < G_NUM) gtmax_ws[b * G_NUM + tid] = __uint_as_float(s_gtmax[tid]);
}

// ---------------------------------------------------------------------------
// Pass B: classification + ordered pos/neg subsampling. One block per batch.
// ---------------------------------------------------------------------------
__global__ void __launch_bounds__(PASSB_THREADS)
passB(const float* __restrict__ gt, const float* __restrict__ anchors,
      const float* __restrict__ max_ov_out, const float* __restrict__ gtmax_ws,
      float* __restrict__ cls_out, int* __restrict__ cls_ws,
      int* __restrict__ posnum_ws) {
    __shared__ float s_gt[G_NUM * 4];
    __shared__ float s_gm[G_NUM];
    __shared__ int   s_cls[N_ANCH];
    __shared__ int   s_cnt[PASSB_THREADS];
    __shared__ int   s_excl[PASSB_THREADS + 1];
    const int b = blockIdx.x;
    const int tid = threadIdx.x;

    if (tid < G_NUM * 4) s_gt[tid] = gt[(size_t)b * G_NUM * 4 + tid];
    if (tid < G_NUM) {
        float gm = gtmax_ws[b * G_NUM + tid];
        s_gm[tid] = (gm == 0.0f) ? 1e-5f : gm;
    }
    __syncthreads();

    const int CH = (N_ANCH + PASSB_THREADS - 1) / PASSB_THREADS;  // 13
    const int n0 = tid * CH;
    const int n1 = (n0 + CH < N_ANCH) ? (n0 + CH) : N_ANCH;

    int pcnt = 0;
    for (int n = n0; n < n1; ++n) {
        float4 an = ((const float4*)anchors)[n];
        float aw = an.z - an.x + 1.0f;
        float ah = an.w - an.y + 1.0f;
        float area_a = aw * ah;
        int keep = 0;
        for (int g = 0; g < G_NUM; ++g) {
            float v = iou_fn(an.x, an.y, an.z, an.w, area_a,
                             s_gt[g * 4 + 0], s_gt[g * 4 + 1],
                             s_gt[g * 4 + 2], s_gt[g * 4 + 3]);
            keep |= (v == s_gm[g]);
        }
        float mv = max_ov_out[(size_t)b * N_ANCH + n];
        int c;
        if (keep) c = 1;
        else if (mv <= THR_LOW) c = 0;
        else if (mv >= THR_HIGH) c = 1;
        else c = -1;
        s_cls[n] = c;
        pcnt += (c == 1);
    }
    s_cnt[tid] = pcnt;
    __syncthreads();
    if (tid == 0) {
        int acc = 0;
        for (int i = 0; i < PASSB_THREADS; ++i) { s_excl[i] = acc; acc += s_cnt[i]; }
        s_excl[PASSB_THREADS] = acc;
    }
    __syncthreads();
    int run = s_excl[tid];
    const int total_pos = s_excl[PASSB_THREADS];
    for (int n = n0; n < n1; ++n)
        if (s_cls[n] == 1) { run++; if (run > POS_CAP) s_cls[n] = -1; }
    __syncthreads();

    int ncnt = 0;
    for (int n = n0; n < n1; ++n) ncnt += (s_cls[n] == 0);
    s_cnt[tid] = ncnt;
    __syncthreads();
    if (tid == 0) {
        int acc = 0;
        for (int i = 0; i < PASSB_THREADS; ++i) { s_excl[i] = acc; acc += s_cnt[i]; }
    }
    __syncthreads();
    run = s_excl[tid];
    for (int n = n0; n < n1; ++n)
        if (s_cls[n] == 0) { run++; if (run > TOTAL_CAP) s_cls[n] = -1; }
    __syncthreads();

    for (int n = n0; n < n1; ++n) {
        int c = s_cls[n];
        cls_ws[(size_t)b * N_ANCH + n] = c;
        int a = n % A_NUM, p = n / A_NUM;  // p = y*25+x
        cls_out[((size_t)b * A_NUM + a) * (SCORE_SZ * SCORE_SZ) + p] = (float)c;
    }
    if (b == B_NUM - 1 && tid == 0)
        posnum_ws[0] = (total_pos > POS_CAP) ? POS_CAP : total_pos;
}

// ---------------------------------------------------------------------------
// Pass C: box-delta transform via WMMA.  Each wave handles 16 anchors:
//   [gcx,gcy,gw,gh](16x4) = matchedGT(16x4) x M(4x4) + const  -> one
//   v_wmma_f32_16x16x32_f16 (A rows = matched boxes in K=0..3, rest zero).
// Then per-lane VALU: dx,dy = (gc - ac)/a_sz ; dw,dh = log(g_sz/a_sz).
// Also writes bw = (cls==1)/pos_num.
// ---------------------------------------------------------------------------
__global__ void __launch_bounds__(256)
passC(const float* __restrict__ gt, const float* __restrict__ anchors,
      const int* __restrict__ arg_ws, const int* __restrict__ cls_ws,
      const int* __restrict__ posnum_ws,
      float* __restrict__ bt_out, float* __restrict__ bw_out) {
    const int b    = blockIdx.y;
    const int wave = threadIdx.x >> 5;
    const int lane = threadIdx.x & 31;
    const int tile = blockIdx.x * 8 + wave;
    const int ntiles = (N_ANCH + 15) / 16;   // 196
    if (tile >= ntiles) return;              // uniform per wave
    const int base = tile * 16;
    const int col  = lane & 15;
    const int half = lane >> 4;

    // A fragment: lanes 0-15 = row m (=lane), K=0..3 carry matched gt coords.
    v16h a;
    #pragma unroll
    for (int i = 0; i < 16; ++i) a[i] = (_Float16)0.0f;
    if (half == 0) {
        int nA = base + lane; if (nA > N_ANCH - 1) nA = N_ANCH - 1;
        int g = arg_ws[(size_t)b * N_ANCH + nA];
        float4 gb = ((const float4*)gt)[b * G_NUM + g];
        a[0] = (_Float16)gb.x; a[1] = (_Float16)gb.y;
        a[2] = (_Float16)gb.z; a[3] = (_Float16)gb.w;
    }
    // B fragment: columns 0..3 = [gcx, gcy, gw, gh] coefficients over gt0..gt3.
    v16h bf;
    #pragma unroll
    for (int i = 0; i < 16; ++i) bf[i] = (_Float16)0.0f;
    if (half == 0) {
        if      (lane == 0) { bf[0] = (_Float16) 0.5f; bf[2] = (_Float16)0.5f; }
        else if (lane == 1) { bf[1] = (_Float16) 0.5f; bf[3] = (_Float16)0.5f; }
        else if (lane == 2) { bf[0] = (_Float16)-1.0f; bf[2] = (_Float16)1.0f; }
        else if (lane == 3) { bf[1] = (_Float16)-1.0f; bf[3] = (_Float16)1.0f; }
    }
    // C: +0.5 for gcx,gcy ; +1.0 for gw,gh (the "+1" width convention).
    float cval = (col < 2) ? 0.5f : ((col < 4) ? 1.0f : 0.0f);
    v8f c;
    #pragma unroll
    for (int i = 0; i < 8; ++i) c[i] = cval;

    v8f d = __builtin_amdgcn_wmma_f32_16x16x32_f16(
        /*neg_a=*/false, a, /*neg_b=*/false, bf,
        /*c_mod=*/(short)0, c, /*reuse_a=*/false, /*reuse_b=*/false);

    if (col < 4) {
        #pragma unroll
        for (int r = 0; r < 8; ++r) {
            int m = r + 8 * half;
            int n2 = base + m;
            if (n2 >= N_ANCH) continue;
            float4 an = ((const float4*)anchors)[n2];
            float aw = an.z - an.x + 1.0f;
            float ah = an.w - an.y + 1.0f;
            float gval = d[r];
            float val;
            if      (col == 0) val = (gval - (an.x + 0.5f * aw)) / aw;
            else if (col == 1) val = (gval - (an.y + 0.5f * ah)) / ah;
            else if (col == 2) val = logf(gval / aw);
            else               val = logf(gval / ah);
            int ai = n2 % A_NUM, p = n2 / A_NUM;
            bt_out[(((size_t)b * 4 + col) * A_NUM + ai) * (SCORE_SZ * SCORE_SZ) + p] = val;
        }
    }
    if (col == 0) {  // lanes 0 and 16 together cover the 16 anchors for bw
        float invp = 1.0f / (float)posnum_ws[0];
        #pragma unroll
        for (int r = 0; r < 8; ++r) {
            int n2 = base + r + 8 * half;
            if (n2 >= N_ANCH) continue;
            int cl = cls_ws[(size_t)b * N_ANCH + n2];
            int ai = n2 % A_NUM, p = n2 / A_NUM;
            bw_out[((size_t)b * A_NUM + ai) * (SCORE_SZ * SCORE_SZ) + p] =
                (cl == 1) ? invp : 0.0f;
        }
    }
}

// ---------------------------------------------------------------------------
extern "C" void kernel_launch(void* const* d_in, const int* in_sizes, int n_in,
                              void* d_out, int out_size, void* d_ws, size_t ws_size,
                              hipStream_t stream) {
    const float* gt      = (const float*)d_in[0];   // (128,64,4)
    const float* anchors = (const float*)d_in[1];   // (3125,4)
    (void)in_sizes; (void)n_in; (void)out_size; (void)ws_size;

    float* out = (float*)d_out;
    // Output layout (floats): cls | bt | bw | max_ov
    float* cls_out    = out;                                   // 128*5*625
    float* bt_out     = out + (size_t)B_NUM * A_NUM * 625;     // 128*4*5*625
    float* bw_out     = bt_out + (size_t)B_NUM * 4 * A_NUM * 625;
    float* max_ov_out = bw_out + (size_t)B_NUM * A_NUM * 625;

    // Workspace layout
    int*   arg_ws    = (int*)d_ws;                               // 128*3125
    float* gtmax_ws  = (float*)(arg_ws + (size_t)B_NUM * N_ANCH);// 128*64
    int*   cls_ws    = (int*)(gtmax_ws + (size_t)B_NUM * G_NUM); // 128*3125
    int*   posnum_ws = cls_ws + (size_t)B_NUM * N_ANCH;          // 1

    passA<<<dim3(B_NUM), dim3(256), 0, stream>>>(gt, anchors, max_ov_out,
                                                 arg_ws, gtmax_ws);
    passB<<<dim3(B_NUM), dim3(PASSB_THREADS), 0, stream>>>(gt, anchors, max_ov_out,
                                                           gtmax_ws, cls_out,
                                                           cls_ws, posnum_ws);
    passC<<<dim3(25, B_NUM), dim3(256), 0, stream>>>(gt, anchors, arg_ws, cls_ws,
                                                     posnum_ws, bt_out, bw_out);
}
